// GNN_51058571215597
// MI455X (gfx1250) — matrix operand order repaired and checked by reading, not compiled, not run on hardware
//
#include <hip/hip_runtime.h>
#include <hip/hip_bf16.h>
#include <math.h>

// ---------------------------------------------------------------------------
// Types for WMMA operands (gfx1250 wave32 WMMA, bf16 -> f32 accumulate)
// ---------------------------------------------------------------------------
typedef __attribute__((ext_vector_type(16))) __bf16 v16bf;
typedef __attribute__((ext_vector_type(8)))  float  v8f;

struct __attribute__((aligned(16))) Q16 { unsigned int x, y, z, w; };
union BF16x16 { v16bf v; Q16 q[2]; };

#define HID 128
#define OUTC 40
#define OUTP 48   // padded output cols for postMP GEMM2 (3 tiles of 16)
#define NGRAPH 64

// ---------------------------------------------------------------------------
// Small helper kernels
// ---------------------------------------------------------------------------
__global__ void f2bf_kernel(const float* __restrict__ in, __bf16* __restrict__ out, int n) {
    int i = blockIdx.x * blockDim.x + threadIdx.x;
    if (i < n) out[i] = (__bf16)in[i];
}

// pad W2 [40,128] -> bf16 [48,128] (rows 40..47 zero)
__global__ void padw2_kernel(const float* __restrict__ w2, __bf16* __restrict__ out) {
    int i = blockIdx.x * blockDim.x + threadIdx.x;
    if (i >= OUTP * HID) return;
    int r = i / HID;
    out[i] = (r < OUTC) ? (__bf16)w2[i] : (__bf16)0.0f;
}

__global__ void deg_kernel(const int* __restrict__ ei, float* __restrict__ deg, int E) {
    int t = blockIdx.x * blockDim.x + threadIdx.x;
    if (t < E) atomicAdd(&deg[ei[E + t]], 1.0f);
}

// one wave per edge: agg[dst] += m[src]  (128 floats; b128 gather, f32 atomics)
__global__ __launch_bounds__(256) void scatter_kernel(const float* __restrict__ m,
                                                      const int* __restrict__ ei,
                                                      float* __restrict__ agg, int E) {
    int wid  = (blockIdx.x * blockDim.x + threadIdx.x) >> 5;
    int lane = threadIdx.x & 31;
    if (wid >= E) return;
    int src = ei[wid];
    int dst = ei[E + wid];
    const float4* mr = (const float4*)(m + (size_t)src * HID);
    float*        ar = agg + (size_t)dst * HID;
    if (wid + 8 < E)  // prefetch a future edge's source row (global_prefetch_b8)
        __builtin_prefetch(m + (size_t)ei[wid + 8] * HID, 0, 0);
    float4 v = mr[lane];
    atomicAdd(&ar[lane * 4 + 0], v.x);
    atomicAdd(&ar[lane * 4 + 1], v.y);
    atomicAdd(&ar[lane * 4 + 2], v.z);
    atomicAdd(&ar[lane * 4 + 3], v.w);
}

// BN (batch stats from atomically-accumulated column sums) + ReLU + cast bf16
__global__ void bn_relu_kernel(const float* __restrict__ h2, const float* __restrict__ stats,
                               const float* __restrict__ gamma, const float* __restrict__ beta,
                               __bf16* __restrict__ out, int nrows) {
    int idx = blockIdx.x * blockDim.x + threadIdx.x;
    if (idx >= nrows * HID) return;
    int c = idx & (HID - 1);
    float invn = 1.0f / (float)nrows;
    float mu  = stats[c] * invn;
    float var = stats[HID + c] * invn - mu * mu;
    float x = h2[idx];
    float y = gamma[c] * (x - mu) * rsqrtf(var + 1e-5f) + beta[c];
    out[idx] = (__bf16)fmaxf(y, 0.0f);
}

__global__ void pool_kernel(const float* __restrict__ y, const int* __restrict__ batch,
                            float* __restrict__ pooled, int n) {
    int idx = blockIdx.x * blockDim.x + threadIdx.x;
    if (idx >= n * OUTC) return;
    int node = idx / OUTC;
    int o    = idx - node * OUTC;
    atomicAdd(&pooled[batch[node] * OUTC + o], y[(size_t)node * OUTP + o]);
}

__global__ void lsm_kernel(const float* __restrict__ pooled, float* __restrict__ out) {
    int g = threadIdx.x;
    if (g >= NGRAPH) return;
    const float* r = pooled + g * OUTC;
    float mx = -INFINITY;
    for (int o = 0; o < OUTC; ++o) mx = fmaxf(mx, r[o]);
    float s = 0.0f;
    for (int o = 0; o < OUTC; ++o) s += expf(r[o] - mx);
    float ls = logf(s) + mx;
    for (int o = 0; o < OUTC; ++o) out[g * OUTC + o] = r[o] - ls;
}

// ---------------------------------------------------------------------------
// Fused WMMA GEMM:  result = X[nrows,128] @ W^T + bias
//   Weights staged global->LDS via CDNA5 async-to-LDS DMA (ASYNCcnt tracked).
//   A fragments (full 128-K row) preloaded; per k-step ALL B fragments are
//   loaded into distinct registers before the wmma chain so the scheduler can
//   use incremental s_wait_dscnt instead of wait-to-zero per MMA.
//   MODE 0: Yf = result                       (fp32, ld = NTILES*16)
//   MODE 1: h2 = result + agg/deg; row-L2-normalize; Yf=h2; accumulate col stats
//   MODE 2: Ybf = bf16(result)
//   MODE 3: Yf = result                       (NTILES=3, ld=48, biasn=40)
// Block: 256 threads = 8 waves; wave = 16 rows x (NTILES*16) cols; K loop = 4x32.
// ---------------------------------------------------------------------------
template <int NTILES, int MODE>
__global__ __launch_bounds__(256) void wmma_gemm_kernel(
    const __bf16* __restrict__ X, const __bf16* __restrict__ W,
    const float* __restrict__ bias, int biasn,
    float* __restrict__ Yf, __bf16* __restrict__ Ybf,
    const float* __restrict__ agg, const float* __restrict__ deg,
    float* __restrict__ stats, int nrows) {
    __shared__ __align__(16) __bf16 Wlds[HID * HID];

    const int tid     = threadIdx.x;
    const int wave    = tid >> 5;
    const int lane    = tid & 31;
    const int lanecol = lane & 15;
    const int khalf   = lane >> 4;           // which K-half this lane holds
    const int rowbase = blockIdx.x * 128 + wave * 16;

    // ---- stage weights into LDS via async DMA (no VGPR round-trip) ----
    {
        const int totalq = (NTILES * 16 * HID) / 8;   // Q16 units (16B each)
#pragma unroll
        for (int i = tid; i < totalq; i += 256) {
            unsigned           ldsa = (unsigned)(uintptr_t)(Wlds + (size_t)i * 8);
            unsigned long long ga   = (unsigned long long)(uintptr_t)(W + (size_t)i * 8);
            asm volatile("global_load_async_to_lds_b128 %0, %1, off"
                         :: "v"(ldsa), "v"(ga) : "memory");
        }
    }

    // ---- preload all A fragments (overlaps with async weight staging) ----
    int arow = rowbase + lanecol;
    if (arow >= nrows) arow = nrows - 1;     // clamp; stores masked later
    const __bf16* xr = X + (size_t)arow * HID;
    BF16x16 A[4];
#pragma unroll
    for (int ks = 0; ks < 4; ++ks) {
        A[ks].q[0] = *(const Q16*)(xr + ks * 32 + khalf * 8);
        A[ks].q[1] = *(const Q16*)(xr + ks * 32 + khalf * 8 + 16);
    }

    // ---- init accumulators with bias (per output column) ----
    v8f acc[NTILES];
#pragma unroll
    for (int t = 0; t < NTILES; ++t) {
        int col = t * 16 + lanecol;
        float bv = (col < biasn) ? bias[col] : 0.0f;
#pragma unroll
        for (int i = 0; i < 8; ++i) acc[t][i] = bv;
    }

    asm volatile("s_wait_asynccnt 0" ::: "memory");
    __syncthreads();

    // ---- main K loop: 4 steps of K=32 ----
    const __bf16* wbase = Wlds + (size_t)lanecol * HID + khalf * 8;
#pragma unroll
    for (int ks = 0; ks < 4; ++ks) {
        // load ALL tile fragments for this k-step into distinct registers
        BF16x16 b[NTILES];
#pragma unroll
        for (int t = 0; t < NTILES; ++t) {
            const __bf16* wn = wbase + (size_t)t * 16 * HID + ks * 32;
            b[t].q[0] = *(const Q16*)(wn);
            b[t].q[1] = *(const Q16*)(wn + 16);
        }
        // back-to-back wmma chain; waits can retire incrementally
#pragma unroll
        for (int t = 0; t < NTILES; ++t)
            acc[t] = __builtin_amdgcn_wmma_f32_16x16x32_bf16(
                false, A[ks].v, false, b[t].v, (short)0, acc[t], false, false);
    }

    const int rhalf = khalf * 8;   // C/D layout: VGPR i -> row i (+8 for upper lanes)

    if (MODE == 1) {
        // ---- add scatter-mean aggregate ----
#pragma unroll
        for (int i = 0; i < 8; ++i) {
            int row = rowbase + rhalf + i;
            if (row < nrows) {
                float inv = 1.0f / fmaxf(deg[row], 1.0f);
#pragma unroll
                for (int t = 0; t < NTILES; ++t)
                    acc[t][i] += agg[(size_t)row * HID + (t * 16 + lanecol)] * inv;
            }
        }
        // ---- row L2 normalize: half-wave (16 lane) reduction per row ----
#pragma unroll
        for (int i = 0; i < 8; ++i) {
            float s = 0.0f;
#pragma unroll
            for (int t = 0; t < NTILES; ++t) s += acc[t][i] * acc[t][i];
            s += __shfl_xor(s, 1, 32);
            s += __shfl_xor(s, 2, 32);
            s += __shfl_xor(s, 4, 32);
            s += __shfl_xor(s, 8, 32);
            float scale = 1.0f / fmaxf(sqrtf(s), 1e-12f);
#pragma unroll
            for (int t = 0; t < NTILES; ++t) acc[t][i] *= scale;
        }
        // ---- store h2 + accumulate column stats for BatchNorm ----
#pragma unroll
        for (int t = 0; t < NTILES; ++t) {
            int col = t * 16 + lanecol;
            float cs = 0.0f, cq = 0.0f;
#pragma unroll
            for (int i = 0; i < 8; ++i) {
                int row = rowbase + rhalf + i;
                if (row < nrows) {
                    float v = acc[t][i];
                    Yf[(size_t)row * HID + col] = v;
                    cs += v;
                    cq += v * v;
                }
            }
            // combine the two half-waves (same column) before hitting atomics
            cs += __shfl_xor(cs, 16, 32);
            cq += __shfl_xor(cq, 16, 32);
            if (khalf == 0) {
                atomicAdd(&stats[col], cs);
                atomicAdd(&stats[HID + col], cq);
            }
        }
    } else if (MODE == 2) {
#pragma unroll
        for (int t = 0; t < NTILES; ++t) {
            int col = t * 16 + lanecol;
#pragma unroll
            for (int i = 0; i < 8; ++i) {
                int row = rowbase + rhalf + i;
                if (row < nrows) Ybf[(size_t)row * HID + col] = (__bf16)acc[t][i];
            }
        }
    } else {  // MODE 0 / 3 : plain fp32 store with ld = NTILES*16
        const int ld = NTILES * 16;
#pragma unroll
        for (int t = 0; t < NTILES; ++t) {
            int col = t * 16 + lanecol;
#pragma unroll
            for (int i = 0; i < 8; ++i) {
                int row = rowbase + rhalf + i;
                if (row < nrows) Yf[(size_t)row * ld + col] = acc[t][i];
            }
        }
    }
}

// ---------------------------------------------------------------------------
// Host orchestration
// ---------------------------------------------------------------------------
extern "C" void kernel_launch(void* const* d_in, const int* in_sizes, int n_in,
                              void* d_out, int out_size, void* d_ws, size_t ws_size,
                              hipStream_t stream) {
    const float* x     = (const float*)d_in[0];
    const int*   ei    = (const int*)d_in[1];   // [2,E]
    const int*   batch = (const int*)d_in[2];   // [N]
    const float* Wl    = (const float*)d_in[3]; // [3,128,128]
    const float* bl    = (const float*)d_in[4];
    const float* Wr    = (const float*)d_in[5];
    const float* br    = (const float*)d_in[6];
    const float* gamma = (const float*)d_in[7];
    const float* beta  = (const float*)d_in[8];
    const float* W1    = (const float*)d_in[9];
    const float* b1    = (const float*)d_in[10];
    const float* W2    = (const float*)d_in[11];
    const float* b2    = (const float*)d_in[12];

    const int N = in_sizes[2];          // nodes
    const int E = in_sizes[1] / 2;      // edges

    // ---- workspace layout ----
    char* ws = (char*)d_ws;
    size_t off = 0;
    auto take = [&](size_t bytes) { char* p = ws + off; off += (bytes + 255) & ~(size_t)255; return p; };
    __bf16* hbuf  = (__bf16*)take((size_t)N * HID * sizeof(__bf16));  // node features (bf16)
    float*  mbuf  = (float*)take((size_t)N * HID * sizeof(float));    // messages / postMP y
    float*  aggbuf= (float*)take((size_t)N * HID * sizeof(float));    // scatter accumulator
    float*  h2buf = (float*)take((size_t)N * HID * sizeof(float));    // pre-BN activations
    float*  degbuf= (float*)take((size_t)N * sizeof(float));
    float*  stats = (float*)take(2 * HID * sizeof(float));
    float*  pooled= (float*)take(NGRAPH * OUTC * sizeof(float));
    __bf16* wr_bf = (__bf16*)take(3 * HID * HID * sizeof(__bf16));
    __bf16* wl_bf = (__bf16*)take(3 * HID * HID * sizeof(__bf16));
    __bf16* w1_bf = (__bf16*)take(HID * HID * sizeof(__bf16));
    __bf16* w2_bf = (__bf16*)take(OUTP * HID * sizeof(__bf16));
    __bf16* tbuf  = (__bf16*)aggbuf;                                  // reuse for postMP hidden
    float*  ybuf  = mbuf;                                             // reuse for postMP logits

    const int TB = 256;
    const int gemm_blocks = (N + 127) / 128;

    // ---- one-time prep: degree + bf16 conversions ----
    hipMemsetAsync(degbuf, 0, (size_t)N * sizeof(float), stream);
    deg_kernel<<<(E + TB - 1) / TB, TB, 0, stream>>>(ei, degbuf, E);
    f2bf_kernel<<<((size_t)N * HID + TB - 1) / TB, TB, 0, stream>>>(x, hbuf, N * HID);
    f2bf_kernel<<<(3 * HID * HID + TB - 1) / TB, TB, 0, stream>>>(Wr, wr_bf, 3 * HID * HID);
    f2bf_kernel<<<(3 * HID * HID + TB - 1) / TB, TB, 0, stream>>>(Wl, wl_bf, 3 * HID * HID);
    f2bf_kernel<<<(HID * HID + TB - 1) / TB, TB, 0, stream>>>(W1, w1_bf, HID * HID);
    padw2_kernel<<<(OUTP * HID + TB - 1) / TB, TB, 0, stream>>>(W2, w2_bf);

    // ---- message-passing layers ----
    for (int l = 0; l < 3; ++l) {
        hipMemsetAsync(aggbuf, 0, (size_t)N * HID * sizeof(float), stream);
        hipMemsetAsync(stats, 0, 2 * HID * sizeof(float), stream);
        // m = h @ Wr^T + br
        wmma_gemm_kernel<8, 0><<<gemm_blocks, TB, 0, stream>>>(
            hbuf, wr_bf + (size_t)l * HID * HID, br + l * HID, HID,
            mbuf, nullptr, nullptr, nullptr, nullptr, N);
        // agg[dst] += m[src]
        scatter_kernel<<<(E + 7) / 8, TB, 0, stream>>>(mbuf, ei, aggbuf, E);
        // h2 = L2norm(agg/deg + h @ Wl^T + bl); column stats
        wmma_gemm_kernel<8, 1><<<gemm_blocks, TB, 0, stream>>>(
            hbuf, wl_bf + (size_t)l * HID * HID, bl + l * HID, HID,
            h2buf, nullptr, aggbuf, degbuf, stats, N);
        // h = relu(BN(h2)) -> bf16
        bn_relu_kernel<<<((size_t)N * HID + TB - 1) / TB, TB, 0, stream>>>(
            h2buf, stats, gamma + l * HID, beta + l * HID, hbuf, N);
    }

    // ---- postMP: Linear(128->128), Linear(128->40) ----
    wmma_gemm_kernel<8, 2><<<gemm_blocks, TB, 0, stream>>>(
        hbuf, w1_bf, b1, HID, nullptr, tbuf, nullptr, nullptr, nullptr, N);
    wmma_gemm_kernel<3, 3><<<gemm_blocks, TB, 0, stream>>>(
        tbuf, w2_bf, b2, OUTC, ybuf, nullptr, nullptr, nullptr, nullptr, N);

    // ---- global_add_pool + log_softmax ----
    hipMemsetAsync(pooled, 0, NGRAPH * OUTC * sizeof(float), stream);
    pool_kernel<<<((size_t)N * OUTC + TB - 1) / TB, TB, 0, stream>>>(ybuf, batch, pooled, N);
    lsm_kernel<<<1, 64, 0, stream>>>(pooled, (float*)d_out);
}
